// STAN_24026047053983
// MI455X (gfx1250) — compile-verified, matrix operand order checked by reading, not run on hardware
//
#include <hip/hip_runtime.h>

// ============================================================================
// MI455X (gfx1250) persistent-decoder kernel.
//
// Design (see analysis): one workgroup owns TILE_B=8 batch elements for all
// 288 timesteps. Per-element attention keys/values (the only large per-step
// operands) are converted to fp16 and parked in LDS (288 KB of the WGP's
// 320 KB) so the 288-step scan never re-touches HBM for them. GRU and MLP
// matmuls (batch-shared weights) run on v_wmma_f32_16x16x32_f16 with weight
// B-fragments persistent in VGPRs; attention (per-sample K=8 matvecs) runs on
// VALU fp32 against fp16 LDS rows with wave32 shuffle softmax reductions.
// ============================================================================

#define BSZ    4096
#define SEQ    144
#define HID    64
#define NH     8
#define DH     8
#define TLEN   288
#define DECSZ  29
#define TILE_B 8
#define NTHR   256

typedef _Float16 v16h __attribute__((ext_vector_type(16)));
typedef float    v8f  __attribute__((ext_vector_type(8)));
typedef _Float16 h8   __attribute__((ext_vector_type(8)));

// ---- LDS byte offsets (total 317,568 B < 320 KB per WGP) -------------------
#define OFF_KEY  0u        // fp16 key  [8e][8h][144s][8d] = 147456 B
#define OFF_VAL  147456u   // fp16 val  [8e][8h][144s][8d] = 147456 B
#define OFF_GI   294912u   // f32 gi    [8][192]           = 6144 B
#define OFF_GH   301056u   // f32 gh    [8][192]           = 6144 B
#define OFF_H32  307200u   // f32 h     [8][64]            = 2048 B
#define OFF_HA   309248u   // fp16 hA   [16][64]           = 2048 B
#define OFF_XA   311296u   // fp16 xA   [16][32]           = 1024 B
#define OFF_CTX  312320u   // fp16 ctxA [16][64]           = 2048 B
#define OFF_MLP  314368u   // f32 mlp   [8][32]            = 1024 B
#define OFF_BIH  315392u   // f32 b_ih  [192]
#define OFF_BHH  316160u   // f32 b_hh  [192]
#define OFF_B1   316928u   // f32 b1    [32]
#define OFF_W2   317056u   // f32 W2    [32]
#define OFF_WK   317184u   // f32 Wk    [64]
#define OFF_BK   317440u   // f32 bk    [8]
#define OFF_B2   317472u   // f32 b2
#define OFF_PREV 317476u   // f32 prev_out [8]
#define SMEM_BYTES 317568u

// CDNA5 16-bit A/B fragment layout (ISA 7.12.2): lanes 0-15 hold K 0-7/16-23,
// lanes 16-31 hold K 8-15/24-31; M (or N for B) = lane%16.
__device__ inline v16h load_A_frag(const _Float16* A, int ldA, int kbase, int lane) {
  v16h a;
  const int m   = lane & 15;
  const int khi = (lane & 16) ? 8 : 0;
#pragma unroll
  for (int t = 0; t < 16; ++t) {
    int k = kbase + khi + (t < 8 ? t : t + 8);
    a[t] = A[m * ldA + k];
  }
  return a;
}

// B[k][n] = W[n][k] (for x @ W^T), fp32 global weights -> fp16 fragment,
// zero-padding K beyond kmax (e.g. 29 -> 32).
__device__ inline v16h load_B_fragT(const float* W, int ldW, int nbase, int kbase,
                                    int kmax, int lane) {
  v16h b;
  const int n   = nbase + (lane & 15);
  const int khi = (lane & 16) ? 8 : 0;
#pragma unroll
  for (int t = 0; t < 16; ++t) {
    int k = kbase + khi + (t < 8 ? t : t + 8);
    b[t] = (k < kmax) ? (_Float16)W[n * ldW + k] : (_Float16)0.0f;
  }
  return b;
}

__device__ inline v8f wmma16(v16h a, v16h b, v8f c) {
  return __builtin_amdgcn_wmma_f32_16x16x32_f16(false, a, false, b, (short)0, c,
                                                false, false);
}

__device__ inline float sigmoidf_fast(float x) { return 1.0f / (1.0f + __expf(-x)); }

__global__ __launch_bounds__(NTHR, 1)
void stan_decoder_kernel(const float* __restrict__ x_dec,
                         const float* __restrict__ enc,
                         const float* __restrict__ hidden,
                         const float* __restrict__ W_ih,
                         const float* __restrict__ W_hh,
                         const float* __restrict__ b_ih,
                         const float* __restrict__ b_hh,
                         const float* __restrict__ Wk,
                         const float* __restrict__ bk,
                         const float* __restrict__ W1,
                         const float* __restrict__ b1,
                         const float* __restrict__ W2,
                         const float* __restrict__ b2,
                         float* __restrict__ out) {
  extern __shared__ char smem[];
  _Float16* key_s = (_Float16*)(smem + OFF_KEY);
  _Float16* val_s = (_Float16*)(smem + OFF_VAL);
  float*    gi_s  = (float*)(smem + OFF_GI);
  float*    gh_s  = (float*)(smem + OFF_GH);
  float*    h32   = (float*)(smem + OFF_H32);
  _Float16* hA    = (_Float16*)(smem + OFF_HA);
  _Float16* xA    = (_Float16*)(smem + OFF_XA);
  _Float16* ctxA  = (_Float16*)(smem + OFF_CTX);
  float*    mlp_s = (float*)(smem + OFF_MLP);
  float*    bih_s = (float*)(smem + OFF_BIH);
  float*    bhh_s = (float*)(smem + OFF_BHH);
  float*    b1_s  = (float*)(smem + OFF_B1);
  float*    w2_s  = (float*)(smem + OFF_W2);
  float*    wk_s  = (float*)(smem + OFF_WK);
  float*    bk_s  = (float*)(smem + OFF_BK);
  float*    b2_s  = (float*)(smem + OFF_B2);
  float*    prev  = (float*)(smem + OFF_PREV);

  const int tid  = threadIdx.x;
  const int wv   = tid >> 5;
  const int lane = tid & 31;
  const int b0   = blockIdx.x * TILE_B;

  // ---------------- one-time init ------------------------------------------
  for (int i = tid; i < 16 * 32; i += NTHR) xA[i] = (_Float16)0.0f;
  for (int i = tid; i < 16 * 64; i += NTHR) { hA[i] = (_Float16)0.0f; ctxA[i] = (_Float16)0.0f; }
  for (int i = tid; i < TILE_B * HID; i += NTHR) {
    int m = i >> 6, c = i & 63;
    float hv = hidden[(size_t)(b0 + m) * HID + c];
    h32[i] = hv;
    hA[m * 64 + c] = (_Float16)hv;
  }
  if (tid < 192) { bih_s[tid] = b_ih[tid]; bhh_s[tid] = b_hh[tid]; }
  if (tid < 64)  wk_s[tid] = Wk[tid];
  if (tid < 32)  { b1_s[tid] = b1[tid]; w2_s[tid] = W2[tid]; }
  if (tid < 8)   { bk_s[tid] = bk[tid]; prev[tid] = 0.0f; }
  if (tid == 0)  b2_s[0] = b2[0];

  // Stage encoder values into LDS fp16 (one-time HBM read for this block):
  // enc[s][b][c] -> val_s[e][h=c/8][s][d=c%8]
  for (int i = tid; i < TILE_B * SEQ * HID; i += NTHR) {
    int e = i / (SEQ * HID);
    int rem = i % (SEQ * HID);
    int s = rem / HID, c = rem % HID;
    float f = enc[((size_t)s * BSZ + (b0 + e)) * HID + c];
    val_s[((e * NH + (c >> 3)) * SEQ + s) * DH + (c & 7)] = (_Float16)f;
  }
  __syncthreads();

  // key = value @ Wk^T + bk, computed in LDS once.
  for (int i = tid; i < TILE_B * NH * SEQ; i += NTHR) {
    int e = i / (NH * SEQ);
    int hh = (i / SEQ) % NH;
    int s = i % SEQ;
    const _Float16* vr = val_s + ((e * NH + hh) * SEQ + s) * DH;
    _Float16* kr = key_s + ((e * NH + hh) * SEQ + s) * DH;
    float vv[8];
#pragma unroll
    for (int d = 0; d < 8; ++d) vv[d] = (float)vr[d];
#pragma unroll
    for (int d = 0; d < 8; ++d) {
      float o = bk_s[d];
#pragma unroll
      for (int d2 = 0; d2 < 8; ++d2) o += vv[d2] * wk_s[d * 8 + d2];
      kr[d] = (_Float16)o;
    }
  }

  // Persistent weight B-fragments in VGPRs.
  v16h wih_f[3], whh_f[3][2];   // waves 0-3: GRU N-tiles {3w..3w+2} of 12
  if (wv < 4) {
#pragma unroll
    for (int j = 0; j < 3; ++j) {
      int nb = (wv * 3 + j) * 16;
      wih_f[j]    = load_B_fragT(W_ih, DECSZ, nb, 0,  DECSZ, lane);
      whh_f[j][0] = load_B_fragT(W_hh, HID,   nb, 0,  HID,   lane);
      whh_f[j][1] = load_B_fragT(W_hh, HID,   nb, 32, HID,   lane);
    }
  }
  v16h w1_f[2][2];              // wave 0: MLP W1 (32x64)
  if (wv == 0) {
#pragma unroll
    for (int nt = 0; nt < 2; ++nt) {
      w1_f[nt][0] = load_B_fragT(W1, HID, nt * 16, 0,  HID, lane);
      w1_f[nt][1] = load_B_fragT(W1, HID, nt * 16, 32, HID, lane);
    }
  }
  __syncthreads();

  // ---------------- 288-step sequential scan -------------------------------
  for (int t = 0; t < TLEN; ++t) {
    // --- stage x_t (fp32 -> fp16), feeding back prev attention output ------
    {
      int m = tid >> 5, k = tid & 31;   // 8 x 32 entries, one per thread
      float xv = 0.0f;
      if (t == 0) {
        if (k < DECSZ) xv = x_dec[(size_t)(b0 + m) * (TLEN + 1) * DECSZ + k];
      } else {
        if (k < DECSZ - 1)       xv = x_dec[((size_t)(b0 + m) * (TLEN + 1) + t) * DECSZ + k];
        else if (k == DECSZ - 1) xv = prev[m];
      }
      xA[m * 32 + k] = (_Float16)xv;
    }
    // prefetch next step's x rows into cache (global_prefetch_b8)
    if (tid < TILE_B && t + 1 < TLEN)
      __builtin_prefetch(x_dec + ((size_t)(b0 + tid) * (TLEN + 1) + (t + 1)) * DECSZ, 0, 1);
    __syncthreads();

    // --- GRU gate GEMMs on WMMA: gi = x@W_ih^T, gh = h@W_hh^T --------------
    if (wv < 4) {
      v16h ax  = load_A_frag(xA, 32, 0,  lane);
      v16h ah0 = load_A_frag(hA, 64, 0,  lane);
      v16h ah1 = load_A_frag(hA, 64, 32, lane);
#pragma unroll
      for (int j = 0; j < 3; ++j) {
        int nb = (wv * 3 + j) * 16;
        v8f z = {};
        v8f ai = wmma16(ax, wih_f[j], z);
        v8f ah = wmma16(ah0, whh_f[j][0], z);
        ah = wmma16(ah1, whh_f[j][1], ah);
        if (!(lane & 16)) {                 // accumulator rows M=0..7 live here
          int n = nb + (lane & 15);
          float bi = bih_s[n], bh = bhh_s[n];
#pragma unroll
          for (int r = 0; r < 8; ++r) {
            gi_s[r * 192 + n] = ai[r] + bi;
            gh_s[r * 192 + n] = ah[r] + bh;
          }
        }
      }
    }
    __syncthreads();

    // --- GRU elementwise + hidden update -----------------------------------
    for (int i = tid; i < TILE_B * HID; i += NTHR) {
      int m = i >> 6, c = i & 63;
      float rr = sigmoidf_fast(gi_s[m * 192 + c]       + gh_s[m * 192 + c]);
      float zz = sigmoidf_fast(gi_s[m * 192 + 64 + c]  + gh_s[m * 192 + 64 + c]);
      float nn = tanhf(gi_s[m * 192 + 128 + c] + rr * gh_s[m * 192 + 128 + c]);
      float hp = h32[m * 64 + c];
      float hn = (1.0f - zz) * nn + zz * hp;
      h32[m * 64 + c] = hn;
      hA[m * 64 + c]  = (_Float16)hn;
    }
    __syncthreads();

    // --- attention: wave e handles element e; lane = head*4 + chunk --------
    {
      const int head = lane >> 2, ck = lane & 3;   // 4 lanes share a head
      float qd[8];
#pragma unroll
      for (int d = 0; d < 8; ++d) qd[d] = h32[wv * 64 + head * 8 + d];
      const _Float16* kbp = key_s + ((wv * NH + head) * SEQ) * DH;
      const _Float16* vbp = val_s + ((wv * NH + head) * SEQ) * DH;

      float lg[SEQ / 4];
      float lmax = -1e30f;
#pragma unroll 4
      for (int i = 0; i < SEQ / 4; ++i) {
        h8 kr = *(const h8*)(kbp + (ck * (SEQ / 4) + i) * DH);
        float l = 0.0f;
#pragma unroll
        for (int d = 0; d < 8; ++d) l += qd[d] * (float)kr[d];
        lg[i] = l;
        lmax = fmaxf(lmax, l);
      }
      lmax = fmaxf(lmax, __shfl_xor(lmax, 1));
      lmax = fmaxf(lmax, __shfl_xor(lmax, 2));

      float se = 0.0f, acc[8] = {0, 0, 0, 0, 0, 0, 0, 0};
#pragma unroll 4
      for (int i = 0; i < SEQ / 4; ++i) {
        float w = __expf(lg[i] - lmax);
        se += w;
        h8 vr = *(const h8*)(vbp + (ck * (SEQ / 4) + i) * DH);
#pragma unroll
        for (int d = 0; d < 8; ++d) acc[d] += w * (float)vr[d];
      }
      se += __shfl_xor(se, 1);
      se += __shfl_xor(se, 2);
#pragma unroll
      for (int d = 0; d < 8; ++d) {
        acc[d] += __shfl_xor(acc[d], 1);
        acc[d] += __shfl_xor(acc[d], 2);
      }
      if (ck == 0) {
        float inv = 1.0f / se;
#pragma unroll
        for (int d = 0; d < 8; ++d)
          ctxA[wv * 64 + head * 8 + d] = (_Float16)(acc[d] * inv);
      }
    }
    __syncthreads();

    // --- MLP head: relu(ctx @ W1^T + b1) on WMMA (wave 0) ------------------
    if (wv == 0) {
      v16h ac0 = load_A_frag(ctxA, 64, 0,  lane);
      v16h ac1 = load_A_frag(ctxA, 64, 32, lane);
#pragma unroll
      for (int nt = 0; nt < 2; ++nt) {
        v8f z = {};
        v8f a = wmma16(ac0, w1_f[nt][0], z);
        a = wmma16(ac1, w1_f[nt][1], a);
        if (!(lane & 16)) {
          int n = nt * 16 + (lane & 15);
          float bb = b1_s[n];
#pragma unroll
          for (int r = 0; r < 8; ++r) mlp_s[r * 32 + n] = fmaxf(a[r] + bb, 0.0f);
        }
      }
    }
    __syncthreads();

    // --- final dot with W2, write output, feed back ------------------------
    if (tid < TILE_B) {
      float o = b2_s[0];
#pragma unroll
      for (int n = 0; n < 32; ++n) o += mlp_s[tid * 32 + n] * w2_s[n];
      out[(size_t)(b0 + tid) * TLEN + t] = o;
      prev[tid] = o;
    }
    __syncthreads();
  }
}

extern "C" void kernel_launch(void* const* d_in, const int* in_sizes, int n_in,
                              void* d_out, int out_size, void* d_ws, size_t ws_size,
                              hipStream_t stream) {
  (void)in_sizes; (void)n_in; (void)d_ws; (void)ws_size; (void)out_size;
  const float* x_dec  = (const float*)d_in[0];
  const float* enc    = (const float*)d_in[1];
  const float* hidden = (const float*)d_in[2];
  const float* W_ih   = (const float*)d_in[3];
  const float* W_hh   = (const float*)d_in[4];
  const float* b_ih   = (const float*)d_in[5];
  const float* b_hh   = (const float*)d_in[6];
  const float* Wk     = (const float*)d_in[7];
  const float* bk     = (const float*)d_in[8];
  const float* W1     = (const float*)d_in[9];
  const float* b1     = (const float*)d_in[10];
  const float* W2     = (const float*)d_in[11];
  const float* b2     = (const float*)d_in[12];
  float* out = (float*)d_out;

  dim3 grid(BSZ / TILE_B), block(NTHR);
  stan_decoder_kernel<<<grid, block, SMEM_BYTES, stream>>>(
      x_dec, enc, hidden, W_ih, W_hh, b_ih, b_hh, Wk, bk, W1, b1, W2, b2, out);
}
